// MultiheadAttentionV2_63522566308456
// MI455X (gfx1250) — matrix operand order repaired and checked by reading, not compiled, not run on hardware
//
#include <hip/hip_runtime.h>
#include <hip/hip_bf16.h>
#include <stdint.h>

// ---------------------------------------------------------------------------
// Types for CDNA5 (gfx1250) wave32 WMMA fragments
// ---------------------------------------------------------------------------
typedef __bf16 bf16;
typedef float  f32x4   __attribute__((ext_vector_type(4)));
typedef float  v8f     __attribute__((ext_vector_type(8)));
typedef bf16   vbf16   __attribute__((ext_vector_type(16)));
typedef bf16   bf16x8  __attribute__((ext_vector_type(8)));
// Exact pointee type of the async-copy builtin params (per clang diagnostic):
typedef int    v4i_gcc __attribute__((vector_size(16)));

#define DEV __device__ __forceinline__

// gfx1250 async global->LDS copy path (ASYNCcnt), with synchronous fallback.
#if __has_builtin(__builtin_amdgcn_global_load_async_to_lds_b128) && \
    __has_builtin(__builtin_amdgcn_s_wait_asynccnt)
#define ASYNC_LDS 1
#else
#define ASYNC_LDS 0
#endif

DEV void cp_b128_to_lds(const bf16* g, bf16* l) {
#if ASYNC_LDS
    // Generic->AS casts via uintptr_t: low 32 bits of a generic LDS address
    // are the LDS offset (aperture rules); global generic == AS1 address.
    __builtin_amdgcn_global_load_async_to_lds_b128(
        (__attribute__((address_space(1))) v4i_gcc*)(uintptr_t)g,
        (__attribute__((address_space(3))) v4i_gcc*)(uintptr_t)l,
        /*offset=*/0, /*cpol=*/0);
#else
    *(bf16x8*)l = *(const bf16x8*)g;   // global_load_b128 + ds_store_b128
#endif
}

DEV void wait_async_copies() {
#if ASYNC_LDS
    __builtin_amdgcn_s_wait_asynccnt(0);
#endif
}

DEV v8f wmma_bf16(vbf16 a, vbf16 b, v8f c) {
    // D(f32 16x16) = A(bf16 16x32) * B(bf16 32x16) + C
    return __builtin_amdgcn_wmma_f32_16x16x32_bf16(
        /*neg_a=*/false, a, /*neg_b=*/false, b,
        /*c_mod=*/(short)0, c, /*reuse_a=*/false, /*reuse_b=*/false);
}

// A fragment (16x32 MxK). Lane l (g = l>>4) holds row l&15:
//   elems 0..7  -> K = g*8 + 0..7
//   elems 8..15 -> K = 16 + g*8 + 0..7
DEV vbf16 load_a_f32(const float* __restrict__ row, int g) {
    const f32x4* p0 = (const f32x4*)(row + g * 8);
    const f32x4* p1 = (const f32x4*)(row + 16 + g * 8);
    f32x4 a0 = p0[0], a1 = p0[1], b0 = p1[0], b1 = p1[1];
    vbf16 r;
#pragma unroll
    for (int i = 0; i < 4; ++i) {
        r[i]      = (bf16)a0[i];
        r[4 + i]  = (bf16)a1[i];
        r[8 + i]  = (bf16)b0[i];
        r[12 + i] = (bf16)b1[i];
    }
    return r;
}

DEV vbf16 load_a_bf16(const bf16* row, int g) {
    bf16x8 c0 = *(const bf16x8*)(row + g * 8);
    bf16x8 c1 = *(const bf16x8*)(row + 16 + g * 8);
    vbf16 r;
#pragma unroll
    for (int i = 0; i < 8; ++i) { r[i] = c0[i]; r[8 + i] = c1[i]; }
    return r;
}

// B fragment (32x16 KxN). Lane l holds column l&15:
//   elems 0..7  -> K = g*8 + 0..7
//   elems 8..15 -> K = 16 + g*8 + 0..7
DEV vbf16 load_b_f32(const float* __restrict__ row, int g) {
    const f32x4* p = (const f32x4*)(row + g * 8);
    const f32x4* q = (const f32x4*)(row + 16 + g * 8);
    f32x4 x0 = p[0], x1 = p[1], x2 = q[0], x3 = q[1];
    vbf16 r;
#pragma unroll
    for (int i = 0; i < 4; ++i) {
        r[i]      = (bf16)x0[i];
        r[4 + i]  = (bf16)x1[i];
        r[8 + i]  = (bf16)x2[i];
        r[12 + i] = (bf16)x3[i];
    }
    return r;
}

DEV vbf16 load_b_bf16(const bf16* row, int g) {
    bf16x8 c0 = *(const bf16x8*)(row + g * 8);
    bf16x8 c1 = *(const bf16x8*)(row + 16 + g * 8);
    vbf16 r;
#pragma unroll
    for (int i = 0; i < 8; ++i) { r[i] = c0[i]; r[8 + i] = c1[i]; }
    return r;
}

// ---------------------------------------------------------------------------
// Y = X @ W^T + bias  (K=512, Nout=512 fixed by problem)
// AMODE: 0 = X is f32, 1 = X is bf16
// SMODE: 0 = store bf16 row-major at [m*ldY + col0 + n]
//        1 = store bf16 V-transposed per head: [((batch*8+head)*64+d)*1024 + s]
//        2 = store f32 row-major (final output)
// ---------------------------------------------------------------------------
template <int AMODE, int SMODE>
__global__ void __launch_bounds__(256) gemm_wmma_kernel(
    const void* __restrict__ Xv, const float* __restrict__ W,
    const float* __restrict__ bias, void* __restrict__ Yv,
    int M, int K, int ldY, int col0)
{
    const int tid  = threadIdx.x;
    const int wave = tid >> 5, lane = tid & 31;
    const int g = lane >> 4, lo = lane & 15;
    const int m_base = blockIdx.x * 128 + wave * 16;
    const int n_base = blockIdx.y * 64;

    const v8f vzero = {0.f, 0.f, 0.f, 0.f, 0.f, 0.f, 0.f, 0.f};
    v8f acc[4];
#pragma unroll
    for (int t = 0; t < 4; ++t) acc[t] = vzero;

    const float* Xf = (const float*)Xv;
    const bf16*  Xb = (const bf16*)Xv;
    const size_t arow = (size_t)(m_base + lo) * K;

    for (int k0 = 0; k0 < K; k0 += 32) {
        if (k0 + 64 < K) {
            if (AMODE == 0) __builtin_prefetch(Xf + arow + k0 + 64, 0, 0);
            else            __builtin_prefetch(Xb + arow + k0 + 64, 0, 0);
        }
        vbf16 a = (AMODE == 0) ? load_a_f32(Xf + arow + k0, g)
                               : load_a_bf16(Xb + arow + k0, g);
#pragma unroll
        for (int t = 0; t < 4; ++t) {
            vbf16 b = load_b_f32(W + (size_t)(n_base + t * 16 + lo) * K + k0, g);
            acc[t] = wmma_bf16(a, b, acc[t]);
        }
    }

#pragma unroll
    for (int t = 0; t < 4; ++t) {
        const int n = n_base + t * 16 + lo;      // column owned by this lane
        const float bb = bias[n];
#pragma unroll
        for (int r = 0; r < 8; ++r) {
            const int m = m_base + r + 8 * g;    // row held in acc vgpr r
            float y = acc[t][r] + bb;
            if (SMODE == 0) {
                ((bf16*)Yv)[(size_t)m * ldY + col0 + n] = (bf16)y;
            } else if (SMODE == 1) {
                const int batch = m >> 10, s = m & 1023;  // S = 1024
                const int head = n >> 6, d = n & 63;      // DV = 64
                ((bf16*)Yv)[(((size_t)(batch * 8 + head) * 64 + d) << 10) + s] = (bf16)y;
            } else {
                ((float*)Yv)[(size_t)m * ldY + n] = y;
            }
        }
    }
}

// ---------------------------------------------------------------------------
// Fused flash attention:  out = softmax(Q K^T / 8 + mask) V   (bf16 WMMA)
//   q : (N, L, 2C) bf16, head slice = 128 channels
//   k : (N, S, 2C) bf16
//   vt: (N, H, 64, S) bf16   (V transposed per head)
//   o : (N, L, C)  bf16
// Block = 128 threads (4 waves), one (batch, head), 64 query rows.
// K/V slabs double-buffered in LDS via async global->LDS copies so the copy
// for slab i+1 overlaps WMMA work on slab i; all 4 waves share the slabs
// (4x less L2 traffic than per-wave loads).
// ---------------------------------------------------------------------------
__global__ void __launch_bounds__(128) flash_attn_kernel(
    const bf16* __restrict__ q, const bf16* __restrict__ k,
    const bf16* __restrict__ vt, const float* __restrict__ mask,
    bf16* __restrict__ o)
{
    const int L = 1024, S = 1024, H = 8;
    const int tid  = threadIdx.x;
    const int wave = tid >> 5, lane = tid & 31;
    const int g = lane >> 4, lo = lane & 15;

    const int tiles = L / 64;
    const int bh   = blockIdx.x / tiles;
    const int tile = blockIdx.x % tiles;
    const int batch = bh >> 3, h = bh & 7;
    const int l0 = tile * 64 + wave * 16;

    __shared__ bf16 kslab[2][32 * 128];   // 2 x 8 KB : K rows s0..s0+31, head slice
    __shared__ bf16 vslab[2][64 * 32];    // 2 x 4 KB : V^T rows d=0..63, 32 s
    __shared__ bf16 plds[4][16 * 32];     // 4 KB     : per-wave P staging
    bf16* myp = plds[wave];

    const bf16* kbase = k + (size_t)(batch * S) * 1024 + h * 128;
    const bf16* vbase = vt + ((size_t)(batch * 8 + h) * 64) * 1024;

    // ---- cooperative slab staging (async on gfx1250, sync fallback) ----
    auto stage = [&](int s0, int b) {
        bf16* kl = kslab[b];
        bf16* vl = vslab[b];
#pragma unroll
        for (int j = 0; j < 4; ++j) {           // 32x128 bf16 = 512 x b128
            const int chunk = tid + 128 * j;
            const int row = chunk >> 4, c8 = (chunk & 15) * 8;
            cp_b128_to_lds(kbase + (size_t)(s0 + row) * 1024 + c8,
                           kl + row * 128 + c8);
        }
#pragma unroll
        for (int j = 0; j < 2; ++j) {           // 64x32 bf16 = 256 x b128
            const int chunk = tid + 128 * j;
            const int row = chunk >> 2, c8 = (chunk & 3) * 8;
            cp_b128_to_lds(vbase + (size_t)row * 1024 + s0 + c8,
                           vl + row * 32 + c8);
        }
    };

    // Q fragments: 16 rows x 128 (4 K-chunks of 32), kept in registers
    const bf16* qrow = q + ((size_t)(batch * L + l0 + lo)) * 1024 + h * 128;
    vbf16 aq[4];
#pragma unroll
    for (int c = 0; c < 4; ++c) aq[c] = load_a_bf16(qrow + c * 32, g);

    const v8f vzero = {0.f, 0.f, 0.f, 0.f, 0.f, 0.f, 0.f, 0.f};
    float m_i[8], l_i[8];
    v8f acc_o[4];
#pragma unroll
    for (int r = 0; r < 8; ++r) { m_i[r] = -1e30f; l_i[r] = 0.f; }
#pragma unroll
    for (int t = 0; t < 4; ++t) acc_o[t] = vzero;

    const float* mbase = mask + (size_t)(batch * H + h) * L * S;

    stage(0, 0);                                   // prologue: fill slab 0

    for (int i = 0; i < S / 32; ++i) {
        const int s0 = i * 32, buf = i & 1;
        wait_async_copies();                       // slab[buf] resident
        __syncthreads();
        if (i + 1 < S / 32) stage(s0 + 32, buf ^ 1);   // overlap next copy

        // ----- scores: two 16x16 tiles over the 32-wide S slab -----
        v8f sc[2] = {vzero, vzero};
#pragma unroll
        for (int sub = 0; sub < 2; ++sub) {
            const bf16* krow = kslab[buf] + (sub * 16 + lo) * 128;
#pragma unroll
            for (int c = 0; c < 4; ++c) {
                vbf16 bk = load_b_bf16(krow + c * 32, g);   // ds_load_b128 x2
                sc[sub] = wmma_bf16(aq[c], bk, sc[sub]);
            }
        }
        // ----- scale (1/8) + additive mask (coalesced b32 stream) -----
#pragma unroll
        for (int sub = 0; sub < 2; ++sub)
#pragma unroll
            for (int r = 0; r < 8; ++r) {
                const int m = r + 8 * g;
                sc[sub][r] = sc[sub][r] * 0.125f +
                             mbase[(size_t)(l0 + m) * S + s0 + sub * 16 + lo];
            }
        // ----- online softmax (rows live in 16-lane halves per acc-vgpr) ----
#pragma unroll
        for (int r = 0; r < 8; ++r) {
            float mx = fmaxf(sc[0][r], sc[1][r]);
            mx = fmaxf(mx, __shfl_xor(mx, 1, 32));
            mx = fmaxf(mx, __shfl_xor(mx, 2, 32));
            mx = fmaxf(mx, __shfl_xor(mx, 4, 32));
            mx = fmaxf(mx, __shfl_xor(mx, 8, 32));
            const float newm  = fmaxf(m_i[r], mx);
            const float alpha = __expf(m_i[r] - newm);
            const float p0 = __expf(sc[0][r] - newm);
            const float p1 = __expf(sc[1][r] - newm);
            float rs = p0 + p1;
            rs += __shfl_xor(rs, 1, 32);
            rs += __shfl_xor(rs, 2, 32);
            rs += __shfl_xor(rs, 4, 32);
            rs += __shfl_xor(rs, 8, 32);
            l_i[r] = l_i[r] * alpha + rs;
            m_i[r] = newm;
            const int m = r + 8 * g;
            myp[m * 32 + lo]      = (bf16)p0;   // P row-major 16x32 in LDS
            myp[m * 32 + 16 + lo] = (bf16)p1;
#pragma unroll
            for (int t = 0; t < 4; ++t) acc_o[t][r] *= alpha;
        }
        // P store -> P load is wave-local: same-wave DS ops complete in order;
        // the explicit wait + memory clobber stops any compiler reordering.
        asm volatile("s_wait_dscnt 0x0" ::: "memory");

        // ----- P (A-frag via LDS) @ V-slab (32 x 64) -----
        vbf16 ap = load_a_bf16(myp + lo * 32, g);
#pragma unroll
        for (int t = 0; t < 4; ++t) {
            vbf16 bv = load_b_bf16(vslab[buf] + (t * 16 + lo) * 32, g);
            acc_o[t] = wmma_bf16(ap, bv, acc_o[t]);
        }
        __syncthreads();   // all waves done with slab[buf] before re-staging
    }

    // ----- normalize and store (N, L, C) bf16 -----
#pragma unroll
    for (int r = 0; r < 8; ++r) {
        const float inv = 1.0f / l_i[r];
        const int m = r + 8 * g;
#pragma unroll
        for (int t = 0; t < 4; ++t)
            o[((size_t)(batch * L + l0 + m)) * 512 + h * 64 + t * 16 + lo] =
                (bf16)(acc_o[t][r] * inv);
    }
}

// ---------------------------------------------------------------------------
// Host-side orchestration
// ---------------------------------------------------------------------------
extern "C" void kernel_launch(void* const* d_in, const int* in_sizes, int n_in,
                              void* d_out, int out_size, void* d_ws, size_t ws_size,
                              hipStream_t stream) {
    (void)in_sizes; (void)n_in; (void)out_size; (void)ws_size;
    const float* q_c  = (const float*)d_in[0];
    const float* q_p  = (const float*)d_in[1];
    const float* k_c  = (const float*)d_in[2];
    const float* k_p  = (const float*)d_in[3];
    const float* v    = (const float*)d_in[4];
    const float* mask = (const float*)d_in[5];
    const float* W_qc = (const float*)d_in[6];
    const float* b_qc = (const float*)d_in[7];
    const float* W_qp = (const float*)d_in[8];
    const float* b_qp = (const float*)d_in[9];
    const float* W_kc = (const float*)d_in[10];
    const float* b_kc = (const float*)d_in[11];
    const float* W_kp = (const float*)d_in[12];
    const float* b_kp = (const float*)d_in[13];
    const float* W_v  = (const float*)d_in[14];
    const float* b_v  = (const float*)d_in[15];
    const float* W_o  = (const float*)d_in[16];
    const float* b_o  = (const float*)d_in[17];

    // Workspace layout (48 MB total):
    bf16* q_bf = (bf16*)d_ws;                    // (N,L,2C)      8M elems, 16 MB
    bf16* k_bf = q_bf + (size_t)8 * 1024 * 1024; // (N,S,2C)      8M elems, 16 MB
    bf16* vt   = k_bf + (size_t)8 * 1024 * 1024; // (N,H,64,S)    4M elems,  8 MB
    bf16* aout = vt   + (size_t)4 * 1024 * 1024; // (N,L,C)       4M elems,  8 MB

    const int M = 8192, K = 512;
    dim3 ggrid(64, 8);   // 64 * 128 = 8192 rows, 8 * 64 = 512 cols

    // Dual projections (f32 in -> bf16 out), concat via column offset
    gemm_wmma_kernel<0, 0><<<ggrid, 256, 0, stream>>>(q_c, W_qc, b_qc, q_bf, M, K, 1024, 0);
    gemm_wmma_kernel<0, 0><<<ggrid, 256, 0, stream>>>(q_p, W_qp, b_qp, q_bf, M, K, 1024, 512);
    gemm_wmma_kernel<0, 0><<<ggrid, 256, 0, stream>>>(k_c, W_kc, b_kc, k_bf, M, K, 1024, 0);
    gemm_wmma_kernel<0, 0><<<ggrid, 256, 0, stream>>>(k_p, W_kp, b_kp, k_bf, M, K, 1024, 512);
    // V projection, stored transposed per head for contiguous PV B-fragments
    gemm_wmma_kernel<0, 1><<<ggrid, 256, 0, stream>>>(v, W_v, b_v, vt, M, K, 0, 0);

    // Fused flash attention: 8 batches * 8 heads * 16 L-tiles of 64 rows
    flash_attn_kernel<<<8 * 8 * 16, 128, 0, stream>>>(q_bf, k_bf, vt, mask, aout);

    // Output projection (bf16 in -> f32 out)
    gemm_wmma_kernel<1, 2><<<ggrid, 256, 0, stream>>>(aout, W_o, b_o, d_out, M, K, 512, 0);
}
